// SelfAttention_59768764891994
// MI455X (gfx1250) — matrix operand order repaired and checked by reading, not compiled, not run on hardware
//
#include <hip/hip_runtime.h>
#include <cstdint>
#include <cstddef>

// ---------- types ----------
typedef __attribute__((ext_vector_type(16))) __bf16        v16bf;
typedef __attribute__((ext_vector_type(8)))  float         v8f;
typedef __attribute__((ext_vector_type(4)))  unsigned int  u32x4v;
typedef __attribute__((ext_vector_type(8)))  unsigned int  u32x8v;

union Frag {
  u32x4v h[2];
  v16bf  v;
};

static constexpr int Bb = 8, Tt = 1024, Dd = 1024, Hh = 16, DHd = 64;

// ---------- WMMA helper ----------
__device__ inline v8f wmma_bf16(v16bf a, v16bf b, v8f c) {
  // D = A x B + C, f32 accumulate, 16x16x32 bf16
  return __builtin_amdgcn_wmma_f32_16x16x32_bf16(false, a, false, b, (short)0, c,
                                                 false, false);
}

// A-matrix 16x32 bf16 fragment: lane holds row M=lane&15,
// elements {kg..kg+7} and {16+kg..16+kg+7}, kg = (lane>>4)*8
__device__ inline v16bf load_afrag(const __bf16* rowptr, int kg) {
  Frag f;
  f.h[0] = *(const u32x4v*)(rowptr + kg);
  f.h[1] = *(const u32x4v*)(rowptr + 16 + kg);
  return f.v;
}

// B-matrix 32x16 bf16 fragment: lane holds column N=lane&15,
// elements {hi*8..hi*8+7} and {16+hi*8..16+hi*8+7}
__device__ inline v16bf load_bfrag(const __bf16* colptr, int hi) {
  Frag f;
  f.h[0] = *(const u32x4v*)(colptr + hi * 8);
  f.h[1] = *(const u32x4v*)(colptr + 16 + hi * 8);
  return f.v;
}

// ---------- CDNA5 async global -> LDS copy (ASYNCcnt-tracked) ----------
// Generic pointers to LDS carry the LDS byte offset in their low 32 bits.
__device__ inline void async_b128(void* lds, const void* g) {
  asm volatile("global_load_async_to_lds_b128 %0, %1, off"
               :: "v"((unsigned int)(uintptr_t)lds),
                  "v"((unsigned long long)(uintptr_t)g)
               : "memory");
}
__device__ inline void wait_async0() {
  asm volatile("s_wait_asynccnt 0x0" ::: "memory");
}

// ---------- CDNA5 Tensor Data Mover: 2D bf16 tile, global -> LDS ----------
// Loads a (tile_d1 rows x 32 cols) bf16 tile whose rows are strideElems apart,
// packed contiguously into LDS at `lds`.  D# groups 0 + 1 only (2-D tensor).
__device__ inline void tdm_load_tile_128x32(void* lds, const void* g,
                                            unsigned strideElems) {
  unsigned long long ga = (unsigned long long)(uintptr_t)g;
  u32x4v g0;
  g0[0] = 1u;                                     // count=1 valid descriptor
  g0[1] = (unsigned)(uintptr_t)lds;               // lds_addr (bytes)
  g0[2] = (unsigned)ga;                           // global_addr[31:0]
  g0[3] = (unsigned)((ga >> 32) & 0x01FFFFFFu)    // global_addr[56:32]
          | (2u << 30);                           // type = 2 ("image")
  u32x8v g1;
  g1[0] = 1u << 16;          // workgroup_mask=0, data_size=1 (2 bytes)
  g1[1] = 0u;                // tensor_dim0 low16 (dim0 = 1<<20) in [31:16]
  g1[2] = 0x10u;             // tensor_dim0 high16 ; tensor_dim1 low16 = 0
  g1[3] = 0x10u              // tensor_dim1 high16 (dim1 = 1<<20)
          | (32u << 16);     // tile_dim0 = 32 elements
  g1[4] = 128u;              // tile_dim1 = 128 rows, tile_dim2 = 0
  g1[5] = strideElems;       // tensor_dim0_stride low 32 (elements)
  g1[6] = 0u;                // stride high16, dim1_stride low16
  g1[7] = 0u;                // dim1_stride high32
  asm volatile("tensor_load_to_lds %0, %1" :: "s"(g0), "s"(g1) : "memory");
}

// ---------- fp32 -> bf16 cast ----------
__global__ __launch_bounds__(256)
void cast_f32_bf16(const float* __restrict__ in, __bf16* __restrict__ out, int n) {
  int i = (blockIdx.x * 256 + threadIdx.x) * 4;
  if (i + 3 < n) {
#pragma unroll
    for (int j = 0; j < 4; ++j) out[i + j] = (__bf16)in[i + j];
  }
}

// ---------- GEMM: C[M,N] = A[M,K] @ W[N,K]^T (both row-major, "NT") ----------
// Block tile 128x128, K-tile 32, 256 threads = 8 waves, wave tile 64x32.
// Tiles are DMA'd by the Tensor Data Mover (wave 0 issues 2 descriptors per
// K-tile), double-buffered in LDS, synchronized with TENSORcnt + barrier.
// MODE 0: scatter bf16 into q/k/v [B,H,T,64].  MODE 1: fp32 -> fout[M,N].
template <int MODE>
__global__ __launch_bounds__(256)
void gemm_bf16_kernel(const __bf16* __restrict__ A, const __bf16* __restrict__ W,
                      int K, int N,
                      __bf16* __restrict__ qout, __bf16* __restrict__ kout,
                      __bf16* __restrict__ vout, float* __restrict__ fout)
{
  __shared__ __bf16 As[2][128][32];
  __shared__ __bf16 Bs[2][128][32];

  const int tid  = threadIdx.x;
  const int w    = tid >> 5, lane = tid & 31;
  const int hi   = lane >> 4, lo = lane & 15, kg = hi * 8;
  const int wm   = w & 1;       // 0..1 -> 64 rows each
  const int wn   = w >> 1;      // 0..3 -> 32 cols each

  const int m0 = blockIdx.y * 128;
  const int n0 = blockIdx.x * 128;

  v8f acc[4][2] = {};

  const int nkt = K >> 5;
  const __bf16* Atile = A + (size_t)m0 * K;   // + kt*32 per tile
  const __bf16* Btile = W + (size_t)n0 * K;

  if (w == 0) {   // TDM issue wave
    tdm_load_tile_128x32(&As[0][0][0], Atile, (unsigned)K);
    tdm_load_tile_128x32(&Bs[0][0][0], Btile, (unsigned)K);
  }

  for (int kt = 0; kt < nkt; ++kt) {
    const int buf = kt & 1;
    if (w == 0) {
      if (kt + 1 < nkt) {   // DMA next tile into the other buffer
        tdm_load_tile_128x32(&As[buf ^ 1][0][0], Atile + (kt + 1) * 32,
                             (unsigned)K);
        tdm_load_tile_128x32(&Bs[buf ^ 1][0][0], Btile + (kt + 1) * 32,
                             (unsigned)K);
        __builtin_amdgcn_s_wait_tensorcnt(2);   // current tile's 2 DMAs done
      } else {
        __builtin_amdgcn_s_wait_tensorcnt(0);
      }
    }
    __syncthreads();

    v16bf af[4], bfr[2];
#pragma unroll
    for (int fm = 0; fm < 4; ++fm)
      af[fm] = load_afrag(&As[buf][wm * 64 + fm * 16 + lo][0], kg);
#pragma unroll
    for (int fn = 0; fn < 2; ++fn)
      bfr[fn] = load_bfrag(&Bs[buf][wn * 32 + fn * 16 + lo][0], hi);

#pragma unroll
    for (int fm = 0; fm < 4; ++fm)
#pragma unroll
      for (int fn = 0; fn < 2; ++fn)
        acc[fm][fn] = wmma_bf16(af[fm], bfr[fn], acc[fm][fn]);
    __syncthreads();
  }

  // epilogue: C element i lives at row i+8*hi, col lo of its 16x16 tile
#pragma unroll
  for (int fm = 0; fm < 4; ++fm) {
#pragma unroll
    for (int fn = 0; fn < 2; ++fn) {
#pragma unroll
      for (int i = 0; i < 8; ++i) {
        int m = m0 + wm * 64 + fm * 16 + i + 8 * hi;
        int n = n0 + wn * 32 + fn * 16 + lo;
        float val = acc[fm][fn][i];
        if constexpr (MODE == 0) {
          int b = m >> 10, t = m & 1023;
          int which = n >> 10, hh = (n >> 6) & 15, d = n & 63;
          __bf16* dst = (which == 0) ? qout : (which == 1) ? kout : vout;
          dst[(size_t)((b * 16 + hh) * 1024 + t) * 64 + d] = (__bf16)val;
        } else {
          fout[(size_t)m * N + n] = val;
        }
      }
    }
  }
}

// ---------- flash attention: one block per (b*H+h, 128 q-rows) ----------
__global__ __launch_bounds__(256)
void attn_kernel(const __bf16* __restrict__ qb, const __bf16* __restrict__ kb,
                 const __bf16* __restrict__ vb, __bf16* __restrict__ ab)
{
  __shared__ __bf16 Ks[32][64];      // 32 keys x 64 dims (row-major)
  __shared__ __bf16 Vt[64][32];      // V transposed: [d][key]
  __shared__ __bf16 Ps[8][16][32];   // per-wave P scratch (A-layout staging)

  const int tid  = threadIdx.x;
  const int w    = tid >> 5, lane = tid & 31;
  const int hi   = lane >> 4, lo = lane & 15, kg = hi * 8;

  const int bh    = blockIdx.x;              // b*16 + h
  const int b     = bh >> 4, h = bh & 15;
  const int qbase = blockIdx.y * 128;

  const __bf16* qhead = qb + (size_t)bh * Tt * DHd;
  const __bf16* khead = kb + (size_t)bh * Tt * DHd;
  const __bf16* vhead = vb + (size_t)bh * Tt * DHd;

  // q fragments for this wave's 16-row strip (K = 64 -> two 16x32 frags)
  const __bf16* qrow = qhead + (size_t)(qbase + w * 16 + lo) * DHd;
  v16bf qf0 = load_afrag(qrow, kg);
  v16bf qf1 = load_afrag(qrow + 32, kg);

  v8f o0 = {}, o1 = {}, o2 = {}, o3 = {};
  v8f mrow, lrow;
#pragma unroll
  for (int i = 0; i < 8; ++i) { mrow[i] = -1e30f; lrow[i] = 0.f; }

  const float scale = 0.125f;   // DH^-0.5

  const int row = tid >> 3;     // 0..31 : key within tile
  const int ch  = tid & 7;      // 0..7  : 8-elem chunk of head dim

  for (int kt = 0; kt < Tt / 32; ++kt) {
    // K tile: direct async global->LDS copy (one b128 per thread)
    async_b128(&Ks[row][ch * 8],
               khead + (size_t)(kt * 32 + row) * DHd + ch * 8);
    // V tile: load + transposed scatter into LDS
    {
      const __bf16* vsrc = vhead + (size_t)(kt * 32 + row) * DHd + ch * 8;
      u32x4v vv = *(const u32x4v*)vsrc;
      const __bf16* ve = (const __bf16*)&vv;
#pragma unroll
      for (int j = 0; j < 8; ++j) Vt[ch * 8 + j][row] = ve[j];
      if (kt + 1 < Tt / 32)
        __builtin_prefetch(vsrc + 32 * DHd, 0, 1);
    }
    wait_async0();
    __syncthreads();

    // scores S[16 q, 32 keys] = q @ k^T : two 16x16 tiles, K-dim = 64
    v8f s0 = {}, s1 = {};
    s0 = wmma_bf16(qf0, load_bfrag(&Ks[lo][0], hi), s0);
    s0 = wmma_bf16(qf1, load_bfrag(&Ks[lo][32], hi), s0);
    s1 = wmma_bf16(qf0, load_bfrag(&Ks[16 + lo][0], hi), s1);
    s1 = wmma_bf16(qf1, load_bfrag(&Ks[16 + lo][32], hi), s1);

    // online softmax per row (row = i + 8*hi; cols spread over 16-lane group)
#pragma unroll
    for (int i = 0; i < 8; ++i) {
      float a = s0[i] * scale;
      float c = s1[i] * scale;
      float t = fmaxf(a, c);
      t = fmaxf(t, __shfl_xor(t, 1, 32));
      t = fmaxf(t, __shfl_xor(t, 2, 32));
      t = fmaxf(t, __shfl_xor(t, 4, 32));
      t = fmaxf(t, __shfl_xor(t, 8, 32));
      float mn = fmaxf(mrow[i], t);
      float f  = __expf(mrow[i] - mn);
      a = __expf(a - mn);
      c = __expf(c - mn);
      float rs = a + c;
      rs += __shfl_xor(rs, 1, 32);
      rs += __shfl_xor(rs, 2, 32);
      rs += __shfl_xor(rs, 4, 32);
      rs += __shfl_xor(rs, 8, 32);
      lrow[i] = lrow[i] * f + rs;
      mrow[i] = mn;
      s0[i] = a; s1[i] = c;
      o0[i] *= f; o1[i] *= f; o2[i] *= f; o3[i] *= f;
    }

    // stage P (16x32 bf16) into wave-private LDS in C-layout, reload as A-frag
#pragma unroll
    for (int i = 0; i < 8; ++i) {
      Ps[w][i + 8 * hi][lo]      = (__bf16)s0[i];
      Ps[w][i + 8 * hi][16 + lo] = (__bf16)s1[i];
    }
    v16bf pf = load_afrag(&Ps[w][lo][0], kg);

    // O[16,64] += P[16,32] @ V[32,64]  (B columns = Vt rows, contiguous)
    o0 = wmma_bf16(pf, load_bfrag(&Vt[0 * 16 + lo][0], hi), o0);
    o1 = wmma_bf16(pf, load_bfrag(&Vt[1 * 16 + lo][0], hi), o1);
    o2 = wmma_bf16(pf, load_bfrag(&Vt[2 * 16 + lo][0], hi), o2);
    o3 = wmma_bf16(pf, load_bfrag(&Vt[3 * 16 + lo][0], hi), o3);

    __syncthreads();
  }

  // normalize and write attn output as bf16 in [B, T, D] layout
#pragma unroll
  for (int i = 0; i < 8; ++i) {
    int    t    = qbase + w * 16 + i + 8 * hi;
    float  inv  = 1.f / lrow[i];
    size_t base = ((size_t)b * Tt + t) * Dd + h * DHd;
    ab[base + 0  + lo] = (__bf16)(o0[i] * inv);
    ab[base + 16 + lo] = (__bf16)(o1[i] * inv);
    ab[base + 32 + lo] = (__bf16)(o2[i] * inv);
    ab[base + 48 + lo] = (__bf16)(o3[i] * inv);
  }
}

// ---------- host launcher ----------
extern "C" void kernel_launch(void* const* d_in, const int* in_sizes, int n_in,
                              void* d_out, int out_size, void* d_ws, size_t ws_size,
                              hipStream_t stream)
{
  (void)in_sizes; (void)n_in; (void)out_size; (void)ws_size;

  const float* x     = (const float*)d_in[0];
  const float* Wqkv  = (const float*)d_in[1];
  const float* Wproj = (const float*)d_in[2];
  float*       out   = (float*)d_out;

  const size_t nX  = (size_t)Bb * Tt * Dd;   // 8M
  const size_t nWq = (size_t)3 * Dd * Dd;    // 3M
  const size_t nWp = (size_t)Dd * Dd;        // 1M

  __bf16* xb  = (__bf16*)d_ws;
  __bf16* wqb = xb  + nX;
  __bf16* wpb = wqb + nWq;
  __bf16* qb  = wpb + nWp;
  __bf16* kb  = qb  + nX;
  __bf16* vb  = kb  + nX;
  __bf16* ab  = vb  + nX;     // total ~88 MB of workspace

  cast_f32_bf16<<<dim3((unsigned)(nX  / 1024)), 256, 0, stream>>>(x,     xb,  (int)nX);
  cast_f32_bf16<<<dim3((unsigned)(nWq / 1024)), 256, 0, stream>>>(Wqkv,  wqb, (int)nWq);
  cast_f32_bf16<<<dim3((unsigned)(nWp / 1024)), 256, 0, stream>>>(Wproj, wpb, (int)nWp);

  // qkv = x @ Wqkv^T  (M=8192, N=3072, K=1024)
  gemm_bf16_kernel<0><<<dim3(3 * Dd / 128, Bb * Tt / 128), 256, 0, stream>>>(
      xb, wqb, Dd, 3 * Dd, qb, kb, vb, nullptr);

  // attention
  attn_kernel<<<dim3(Bb * Hh, Tt / 128), 256, 0, stream>>>(qb, kb, vb, ab);

  // out = attn_out @ Wproj^T  (M=8192, N=1024, K=1024), fp32 result
  gemm_bf16_kernel<1><<<dim3(Dd / 128, Bb * Tt / 128), 256, 0, stream>>>(
      ab, wpb, Dd, Dd, nullptr, nullptr, nullptr, out);
}
